// MultiHeadAttention_41446434406615
// MI455X (gfx1250) — matrix operand order repaired and checked by reading, not compiled, not run on hardware
//
#include <hip/hip_runtime.h>
#include <hip/hip_bf16.h>
#include <math.h>

typedef __attribute__((ext_vector_type(16))) _Float16 v16h;
typedef __attribute__((ext_vector_type(8)))  _Float16 v8h;
typedef __attribute__((ext_vector_type(8)))  float    v8f;

constexpr int BB = 2, SS = 2048, DD = 1024, HH = 16, DHH = 64;

// ---------------------------------------------------------------------------
// WMMA helpers (v_wmma_f32_16x16x32_f16, wave32)
// ---------------------------------------------------------------------------
__device__ __forceinline__ v8f wmma_f16(v16h a, v16h b, v8f c) {
  // (neg_a, A, neg_b, B, c_mod, C, reuse_a, reuse_b)
  return __builtin_amdgcn_wmma_f32_16x16x32_f16(false, a, false, b, (short)0, c,
                                                false, false);
}

// A-fragment (16x32, M x K), source row-major with K contiguous, ld in elems.
// lane r = lane&15 -> row r ; h = lane>>4 selects K groups {8h..8h+7, 16+8h..}
__device__ __forceinline__ v16h load_frag_a(const _Float16* tile, int ld) {
  const int lane = threadIdx.x & 31;
  const int r = lane & 15;
  const int h = lane >> 4;
  const _Float16* row = tile + (size_t)r * ld + h * 8;
  v8h lo = *reinterpret_cast<const v8h*>(row);
  v8h hi = *reinterpret_cast<const v8h*>(row + 16);
  v16h f;
#pragma unroll
  for (int i = 0; i < 8; ++i) { f[i] = lo[i]; f[8 + i] = hi[i]; }
  return f;
}

// B-fragment (32x16, K x N) where memory holds Bt[n][k] row-major, K contig
// (i.e. the "TN" gemm operand). lane n = lane&15 ; g = lane>>4 -> K groups {8g..8g+7, 16+8g..}.
__device__ __forceinline__ v16h load_frag_b(const _Float16* tile, int ld) {
  const int lane = threadIdx.x & 31;
  const int nn = lane & 15;
  const int g = lane >> 4;
  const _Float16* row = tile + (size_t)nn * ld + g * 8;
  v8h lo = *reinterpret_cast<const v8h*>(row);
  v8h hi = *reinterpret_cast<const v8h*>(row + 16);
  v16h f;
#pragma unroll
  for (int i = 0; i < 8; ++i) { f[i] = lo[i]; f[8 + i] = hi[i]; }
  return f;
}

// ---------------------------------------------------------------------------
// CDNA5 async global->LDS copy (ASYNCcnt path, CDNA5 ISA)
// ---------------------------------------------------------------------------
__device__ __forceinline__ uint32_t lds_byte_off(const void* p) {
  // generic -> addrspace(3) cast; ptrtoint of an LDS pointer yields the
  // 32-bit offset within the wave's LDS allocation (what VDST expects).
  return (uint32_t)(uintptr_t)(__attribute__((address_space(3))) const void*)p;
}

__device__ __forceinline__ void async_copy_b128(uint32_t lds_off,
                                                const _Float16* gaddr) {
  asm volatile("global_load_async_to_lds_b128 %0, %1, off"
               :
               : "v"(lds_off), "v"(gaddr)
               : "memory");
}

__device__ __forceinline__ void wait_async0() {
  asm volatile("s_wait_asynccnt 0" ::: "memory");
}

// ---------------------------------------------------------------------------
// elementwise f32 -> f16
// ---------------------------------------------------------------------------
__global__ void f32_to_f16(const float* __restrict__ in, _Float16* __restrict__ out,
                           int n) {
  int i = blockIdx.x * blockDim.x + threadIdx.x;
  if (i < n) out[i] = (_Float16)in[i];
}

// ---------------------------------------------------------------------------
// C[M,N] (f32) = A_h[M,K] * B_h[N,K]^T + bias[N]     (x @ W.T + b)
// block: 256 threads = 8 waves; block tile 64(M) x 64(N), K-step 64.
// A/B 64x64 f16 tiles staged in LDS via global_load_async_to_lds_b128;
// wave (w>>2) owns 32 M-rows (two 16x16 tiles), (w&3) owns a 16-col chunk.
// ---------------------------------------------------------------------------
__global__ __launch_bounds__(256) void gemm_tn_bias(
    const _Float16* __restrict__ A, const _Float16* __restrict__ Bm,
    const float* __restrict__ bias, float* __restrict__ C, int M, int N, int K) {
  __shared__ __align__(16) _Float16 sA[64 * 64];
  __shared__ __align__(16) _Float16 sB[64 * 64];
  const int t = threadIdx.x;
  const int w = t >> 5;
  const int lane = t & 31;
  const int mblk = blockIdx.y * 64;
  const int nblk = blockIdx.x * 64;
  const int m0 = (w >> 2) * 32;  // local row of this wave's first tile
  const int n0 = (w & 3) * 16;   // local col chunk

  // staging map: tile is 64 rows x 64 halfs = 512 x 16B chunks, 2 per thread
  const int rA0 = t >> 3;        // rows 0..31
  const int rA1 = rA0 + 32;      // rows 32..63
  const int ch = (t & 7) * 8;    // k-offset (halfs) of this thread's chunk
  const uint32_t offA = lds_byte_off(sA);
  const uint32_t offB = lds_byte_off(sB);
  const uint32_t dA0 = offA + (uint32_t)(rA0 * 64 + ch) * 2;
  const uint32_t dA1 = offA + (uint32_t)(rA1 * 64 + ch) * 2;
  const uint32_t dB0 = offB + (uint32_t)(rA0 * 64 + ch) * 2;
  const uint32_t dB1 = offB + (uint32_t)(rA1 * 64 + ch) * 2;

  v8f acc0 = {};
  v8f acc1 = {};
  for (int kk = 0; kk < K; kk += 64) {
    async_copy_b128(dA0, A + (size_t)(mblk + rA0) * K + kk + ch);
    async_copy_b128(dA1, A + (size_t)(mblk + rA1) * K + kk + ch);
    async_copy_b128(dB0, Bm + (size_t)(nblk + rA0) * K + kk + ch);
    async_copy_b128(dB1, Bm + (size_t)(nblk + rA1) * K + kk + ch);
    wait_async0();
    __syncthreads();
#pragma unroll
    for (int ks = 0; ks < 64; ks += 32) {
      v16h bf = load_frag_b(sB + n0 * 64 + ks, 64);
      v16h a0 = load_frag_a(sA + m0 * 64 + ks, 64);
      v16h a1 = load_frag_a(sA + (m0 + 16) * 64 + ks, 64);
      acc0 = wmma_f16(a0, bf, acc0);
      acc1 = wmma_f16(a1, bf, acc1);
    }
    __syncthreads();
  }
  const int n = nblk + n0 + (lane & 15);
  const int h = lane >> 4;
  const float bn = bias[n];
#pragma unroll
  for (int j = 0; j < 8; ++j) {
    C[(size_t)(mblk + m0 + j + 8 * h) * N + n] = acc0[j] + bn;
    C[(size_t)(mblk + m0 + 16 + j + 8 * h) * N + n] = acc1[j] + bn;
  }
}

// ---------------------------------------------------------------------------
// per-head LayerNorm over last-64 (biased variance, eps=1e-5), f32 -> f16
// one thread per (b,s,h) row of 64 contiguous elements
// ---------------------------------------------------------------------------
__global__ void ln_head64(const float* __restrict__ in, _Float16* __restrict__ out,
                          int rows) {
  int r = blockIdx.x * blockDim.x + threadIdx.x;
  if (r >= rows) return;
  const float* p = in + (size_t)r * 64;
  float s = 0.f, s2 = 0.f;
#pragma unroll
  for (int i = 0; i < 64; ++i) { float v = p[i]; s += v; s2 += v * v; }
  float mu = s * (1.0f / 64.0f);
  float var = s2 * (1.0f / 64.0f) - mu * mu;
  float rstd = rsqrtf(var + 1e-5f);
  _Float16* o = out + (size_t)r * 64;
#pragma unroll
  for (int i = 0; i < 64; ++i) o[i] = (_Float16)((p[i] - mu) * rstd);
}

// ---------------------------------------------------------------------------
// v[B,S,H,DH] (f32) -> vt[B,H,DH,S] (f16)   (so P*V gemm has K=keys contiguous)
// ---------------------------------------------------------------------------
__global__ void v_to_f16_T(const float* __restrict__ v, _Float16* __restrict__ vt,
                           int total) {
  int i = blockIdx.x * blockDim.x + threadIdx.x;
  if (i >= total) return;
  int s = i & (SS - 1);
  int t = i >> 11;           // / SS
  int d = t & (DHH - 1);
  int t2 = t >> 6;           // / DHH
  int h = t2 & (HH - 1);
  int b = t2 >> 4;           // / HH
  vt[i] = (_Float16)v[(((size_t)b * SS + s) * HH + h) * DHH + d];
}

// ---------------------------------------------------------------------------
// causal flash attention, one wave per 16-query tile.
// Q,K: [B,S,H,DH] f16   Vt: [B,H,DH,S] f16   O: [B,S,H,DH] f16
// block = 128 threads = 4 waves handling 4 consecutive q-tiles of one (b,h).
// ---------------------------------------------------------------------------
__global__ __launch_bounds__(128) void attn_causal(
    const _Float16* __restrict__ Q, const _Float16* __restrict__ Kh,
    const _Float16* __restrict__ Vt, _Float16* __restrict__ O) {
  __shared__ __align__(16) _Float16 pbuf[4][16 * 32];
  const int w = threadIdx.x >> 5;
  const int lane = threadIdx.x & 31;
  const int n = lane & 15;
  const int hh = lane >> 4;
  const int b = blockIdx.z;
  const int hd = blockIdx.y;
  const int qb = blockIdx.x * 64 + w * 16;

  const _Float16* Qbase = Q + ((size_t)b * SS + qb) * DD + hd * DHH;
  const _Float16* Kbase = Kh + (size_t)b * SS * DD + hd * DHH;
  const _Float16* Vbase = Vt + ((size_t)b * HH + hd) * DHH * SS;
  _Float16* lds = pbuf[w];

  const v16h qa0 = load_frag_a(Qbase, DD);        // d = 0..31
  const v16h qa1 = load_frag_a(Qbase + 32, DD);   // d = 32..63

  v8f o0 = {}, o1 = {}, o2 = {}, o3 = {};
  float mrow[8], lrow[8];
#pragma unroll
  for (int j = 0; j < 8; ++j) { mrow[j] = -INFINITY; lrow[j] = 0.f; }

  const int kend = qb + 16;                       // causal: keys <= qb+15
  for (int kk = 0; kk < kend; kk += 32) {
    // scores S[16q x 32k] = Q(16x64) * K(keys kk..kk+31)^T, scaled 1/sqrt(64)
    v16h kf0a = load_frag_b(Kbase + (size_t)kk * DD, DD);
    v16h kf0b = load_frag_b(Kbase + (size_t)kk * DD + 32, DD);
    v16h kf1a = load_frag_b(Kbase + (size_t)(kk + 16) * DD, DD);
    v16h kf1b = load_frag_b(Kbase + (size_t)(kk + 16) * DD + 32, DD);
    v8f s0 = {}, s1 = {};
    s0 = wmma_f16(qa0, kf0a, s0);
    s0 = wmma_f16(qa1, kf0b, s0);
    s1 = wmma_f16(qa0, kf1a, s1);
    s1 = wmma_f16(qa1, kf1b, s1);

    const int c0 = kk + n;
    const int c1 = kk + 16 + n;
#pragma unroll
    for (int j = 0; j < 8; ++j) {
      const int qrow = qb + j + 8 * hh;
      float a = (c0 <= qrow) ? s0[j] * 0.125f : -INFINITY;
      float c = (c1 <= qrow) ? s1[j] * 0.125f : -INFINITY;
      float rm = fmaxf(a, c);
#pragma unroll
      for (int m = 1; m < 16; m <<= 1) rm = fmaxf(rm, __shfl_xor(rm, m, 32));
      const float mnew = fmaxf(mrow[j], rm);
      const float alpha = __expf(mrow[j] - mnew);
      const float p0 = __expf(a - mnew);
      const float p1 = __expf(c - mnew);
      float rs = p0 + p1;
#pragma unroll
      for (int m = 1; m < 16; m <<= 1) rs += __shfl_xor(rs, m, 32);
      lrow[j] = lrow[j] * alpha + rs;
      mrow[j] = mnew;
      o0[j] *= alpha; o1[j] *= alpha; o2[j] *= alpha; o3[j] *= alpha;
      const int mrel = j + 8 * hh;                 // C-layout row this lane owns
      lds[mrel * 32 + n] = (_Float16)p0;           // P tile row-major 16x32
      lds[mrel * 32 + 16 + n] = (_Float16)p1;
    }
    asm volatile("s_wait_dscnt 0" ::: "memory");   // wave-local LDS relayout

    const v16h pa = load_frag_a(lds, 32);          // P as A-fragment (f16)
    v16h vf0 = load_frag_b(Vbase + (size_t)0 * SS + kk, SS);
    v16h vf1 = load_frag_b(Vbase + (size_t)16 * SS + kk, SS);
    v16h vf2 = load_frag_b(Vbase + (size_t)32 * SS + kk, SS);
    v16h vf3 = load_frag_b(Vbase + (size_t)48 * SS + kk, SS);
    o0 = wmma_f16(pa, vf0, o0);
    o1 = wmma_f16(pa, vf1, o1);
    o2 = wmma_f16(pa, vf2, o2);
    o3 = wmma_f16(pa, vf3, o3);
  }

  _Float16* Ob = O + ((size_t)b * SS + qb) * DD + hd * DHH;
#pragma unroll
  for (int j = 0; j < 8; ++j) {
    const float inv = 1.0f / lrow[j];
    const int m = j + 8 * hh;
    Ob[(size_t)m * DD + 0 + n]  = (_Float16)(o0[j] * inv);
    Ob[(size_t)m * DD + 16 + n] = (_Float16)(o1[j] * inv);
    Ob[(size_t)m * DD + 32 + n] = (_Float16)(o2[j] * inv);
    Ob[(size_t)m * DD + 48 + n] = (_Float16)(o3[j] * inv);
  }
}

// ---------------------------------------------------------------------------
extern "C" void kernel_launch(void* const* d_in, const int* in_sizes, int n_in,
                              void* d_out, int out_size, void* d_ws, size_t ws_size,
                              hipStream_t stream) {
  const float* x  = (const float*)d_in[0];
  // d_in[1] = mask (causal, known statically) -> ignored
  const float* Wq = (const float*)d_in[2];
  const float* bq = (const float*)d_in[3];
  const float* Wk = (const float*)d_in[4];
  const float* bk = (const float*)d_in[5];
  const float* Wv = (const float*)d_in[6];
  const float* bv = (const float*)d_in[7];
  const float* Wo = (const float*)d_in[8];
  const float* bo = (const float*)d_in[9];
  float* out = (float*)d_out;

  const size_t M = (size_t)BB * SS;  // 4096 rows

  char* ws = (char*)d_ws;
  size_t off = 0;
  auto alloc = [&](size_t bytes) -> void* {
    void* p = ws + off;
    off += (bytes + 255) & ~(size_t)255;
    return p;
  };
  _Float16* xh  = (_Float16*)alloc(M * DD * 2);
  _Float16* Wqh = (_Float16*)alloc((size_t)DD * DD * 2);
  _Float16* Wkh = (_Float16*)alloc((size_t)DD * DD * 2);
  _Float16* Wvh = (_Float16*)alloc((size_t)DD * DD * 2);
  _Float16* Woh = (_Float16*)alloc((size_t)DD * DD * 2);
  float*    qf  = (float*)alloc(M * DD * 4);
  float*    kf  = (float*)alloc(M * DD * 4);
  float*    vf  = (float*)alloc(M * DD * 4);
  _Float16* Qh  = (_Float16*)alloc(M * DD * 2);
  _Float16* Kh2 = (_Float16*)alloc(M * DD * 2);
  _Float16* VtT = (_Float16*)alloc(M * DD * 2);
  _Float16* att = (_Float16*)qf;  // alias: q-f32 dead after LN

  auto cdiv = [](int a, int b) { return (a + b - 1) / b; };
  const int nXD = (int)(M * DD);
  const int nWW = DD * DD;

  // 1) precision convert
  f32_to_f16<<<cdiv(nXD, 256), 256, 0, stream>>>(x, xh, nXD);
  f32_to_f16<<<cdiv(nWW, 256), 256, 0, stream>>>(Wq, Wqh, nWW);
  f32_to_f16<<<cdiv(nWW, 256), 256, 0, stream>>>(Wk, Wkh, nWW);
  f32_to_f16<<<cdiv(nWW, 256), 256, 0, stream>>>(Wv, Wvh, nWW);
  f32_to_f16<<<cdiv(nWW, 256), 256, 0, stream>>>(Wo, Woh, nWW);

  // 2) q/k/v projections (WMMA + async LDS staging), f32 out
  dim3 gg(DD / 64, (int)(M / 64));
  gemm_tn_bias<<<gg, 256, 0, stream>>>(xh, Wqh, bq, qf, (int)M, DD, DD);
  gemm_tn_bias<<<gg, 256, 0, stream>>>(xh, Wkh, bk, kf, (int)M, DD, DD);
  gemm_tn_bias<<<gg, 256, 0, stream>>>(xh, Wvh, bv, vf, (int)M, DD, DD);

  // 3) per-head LayerNorm on q,k ; transpose+convert v
  const int rows = (int)(M * HH);
  ln_head64<<<cdiv(rows, 256), 256, 0, stream>>>(qf, Qh, rows);
  ln_head64<<<cdiv(rows, 256), 256, 0, stream>>>(kf, Kh2, rows);
  const int vtot = (int)(M * DD);
  v_to_f16_T<<<cdiv(vtot, 256), 256, 0, stream>>>(vf, VtT, vtot);

  // 4) causal flash attention (WMMA)
  dim3 ga(SS / 64, HH, BB);
  attn_causal<<<ga, 128, 0, stream>>>(Qh, Kh2, VtT, att);

  // 5) output projection (WMMA), f32 out with bias
  gemm_tn_bias<<<gg, 256, 0, stream>>>(att, Woh, bo, out, (int)M, DD, DD);
}
